// ODE_RNN_39728447488707
// MI455X (gfx1250) — compile-verified, matrix operand order checked by reading, not run on hardware
//
#include <hip/hip_runtime.h>
#include <hip/hip_bf16.h>
#include <stdint.h>
#include <stddef.h>

// ---------------- problem constants ----------------
#define BATCH   8192
#define KH      1024      // K_H
#define KT      64        // K_T
#define KOUT    64        // K_OUT
#define KXT     512       // K_EXP + K_IN
#define FHID    2176      // (K_H + K_T) * 2
#define HTI     1088      // K_H + K_T
#define STLD    1088      // state / output leading dim (K_OUT + K_H)

// ---------------- WMMA vector types ----------------
typedef __attribute__((ext_vector_type(16))) __bf16 v16bf;
typedef __attribute__((ext_vector_type(8)))  float  v8f;
typedef int v4i_t __attribute__((vector_size(16)));   // matches builtin param type

// hardware f32 -> bf16 convert (v_cvt path, RNE)
__device__ __forceinline__ unsigned short f32_to_bf16(float f) {
    union { __bf16 b; unsigned short u; } c;
    c.b = (__bf16)f;
    return c.u;
}
__device__ __forceinline__ float sigmoidf(float x) {
    return 1.0f / (1.0f + __expf(-x));
}

// ---------------- async global->LDS copy (CDNA5) ----------------
#if defined(__HIP_DEVICE_COMPILE__) && __has_builtin(__builtin_amdgcn_global_load_async_to_lds_b128)
#define HAS_ASYNC 1
#else
#define HAS_ASYNC 0
#endif

#if defined(__HIP_DEVICE_COMPILE__) && __has_builtin(__builtin_amdgcn_s_wait_asynccnt)
#define WAIT_ASYNC() __builtin_amdgcn_s_wait_asynccnt(0)
#else
#define WAIT_ASYNC() asm volatile("s_wait_asynccnt 0" ::: "memory")
#endif

#if HAS_ASYNC
// copy 32 contiguous bytes (2 x b128) global -> LDS asynchronously
__device__ __forceinline__ void async_cp32(const unsigned short* g, unsigned short* l) {
    auto gp = (__attribute__((address_space(1))) v4i_t*)((unsigned long long)g);
    auto lp = (__attribute__((address_space(3))) v4i_t*)(l);
    __builtin_amdgcn_global_load_async_to_lds_b128(gp, lp, 0, 0);
    __builtin_amdgcn_global_load_async_to_lds_b128(gp, lp, 16, 0);
}
#endif

// ---------------- tiled bf16 WMMA GEMM ----------------
// C[M,N] = A[M,K] @ W[N,K]^T + bias, fused epilogues.
// A, W: bf16 row-major (both pre-packed; W padded in N so no load guards).
// 256 threads = 8 waves; block tile 128x128; K-step 32; double-buffered LDS;
// each wave computes a 32x64 sub-tile as 2x4 WMMA 16x16x32 tiles.
#define BM 128
#define BN 128
#define BK 32
#define LDSW 40   // padded ushort stride per row (32 data + 8 pad)

// EPI: 0 = plain f32 out
//      1 = tanh, write bf16 out only           (GEMM1)
//      2 = ht + val*dt, write f32 + bf16       (GEMM2)
//      3 = y_type mask vs yt, write f32        (GEMM5)
template<int EPI>
__global__ void __launch_bounds__(256)
gemm_bf16_wmma(const unsigned short* __restrict__ A,
               const unsigned short* __restrict__ W,
               const float* __restrict__ bias,
               int N, int K,
               float* __restrict__ outF, int ldOutF,
               unsigned short* __restrict__ outB, int ldOutB,
               const float* __restrict__ aux0, int ldAux0,   // ht (EPI=2) / yt (EPI=3)
               const float* __restrict__ dt,                 // (B,1), EPI=2
               const int* __restrict__ ytype)                // (N,),  EPI=3
{
    __shared__ __align__(16) unsigned short sA[2][BM * LDSW];
    __shared__ __align__(16) unsigned short sB[2][BN * LDSW];

    const int tid  = threadIdx.x;
    const int lane = tid & 31;
    const int wid  = tid >> 5;
    const int wm   = wid & 3;    // 4 waves along M -> 32-row slices
    const int wn   = wid >> 2;   // 2 waves along N -> 64-col slices

    const int mbase = blockIdx.y * BM;
    const int nbase = blockIdx.x * BN;

    const int ldr = tid >> 1;    // 0..127: tile row for cooperative loads
    const int ldh = tid & 1;     // which 16-element K half

    // per-thread source/dest for tile staging (32B per thread per tile)
    const unsigned short* gA = A + (size_t)(mbase + ldr) * K + ldh * 16;
    const unsigned short* gW = W + (size_t)(nbase + ldr) * K + ldh * 16;
    const int lofs = ldr * LDSW + ldh * 16;

    const v8f vzero = {0.f,0.f,0.f,0.f,0.f,0.f,0.f,0.f};
    v8f acc[2][4];
    #pragma unroll
    for (int i = 0; i < 2; ++i)
        #pragma unroll
        for (int j = 0; j < 4; ++j) acc[i][j] = vzero;

    const int mloc  = lane & 15;  // row within 16x16 tile (A) / col (B)
    const int khalf = lane >> 4;  // K-half select per ISA 16-bit layout
    const int nsteps = K >> 5;    // K / BK

#if HAS_ASYNC
    // prologue: fill buffer 0
    async_cp32(gA, &sA[0][lofs]);
    async_cp32(gW, &sB[0][lofs]);
    WAIT_ASYNC();
    __syncthreads();
#else
    uint4 ra0, ra1, rw0, rw1;
    {
        const uint4* pa = (const uint4*)gA;       ra0 = pa[0]; ra1 = pa[1];
        const uint4* pw = (const uint4*)gW;       rw0 = pw[0]; rw1 = pw[1];
        uint4* qa = (uint4*)&sA[0][lofs];         qa[0] = ra0; qa[1] = ra1;
        uint4* qw = (uint4*)&sB[0][lofs];         qw[0] = rw0; qw[1] = rw1;
    }
    __syncthreads();
#endif

    for (int s = 0; s < nsteps; ++s) {
        const int cur = s & 1;
        const int nxt = cur ^ 1;

        // kick off next stage while computing this one
        if (s + 1 < nsteps) {
            const int kb = (s + 1) * BK;
#if HAS_ASYNC
            async_cp32(gA + kb, &sA[nxt][lofs]);
            async_cp32(gW + kb, &sB[nxt][lofs]);
#else
            const uint4* pa = (const uint4*)(gA + kb); ra0 = pa[0]; ra1 = pa[1];
            const uint4* pw = (const uint4*)(gW + kb); rw0 = pw[0]; rw1 = pw[1];
#endif
        }

        // ---- gather fragments per ISA 7.12.2 16-bit A/B layout ----
        union Frag { v16bf v; uint4 q[2]; };
        Frag af[2], bfg[4];
        #pragma unroll
        for (int mt = 0; mt < 2; ++mt) {
            const int rb = (wm * 32 + mt * 16 + mloc) * LDSW;
            af[mt].q[0] = *(const uint4*)&sA[cur][rb + khalf * 8];
            af[mt].q[1] = *(const uint4*)&sA[cur][rb + 16 + khalf * 8];
        }
        #pragma unroll
        for (int nt = 0; nt < 4; ++nt) {
            const int rb = (wn * 64 + nt * 16 + mloc) * LDSW;
            bfg[nt].q[0] = *(const uint4*)&sB[cur][rb + khalf * 8];
            bfg[nt].q[1] = *(const uint4*)&sB[cur][rb + 16 + khalf * 8];
        }
        // ---- 8 WMMAs per wave per K-step ----
        #pragma unroll
        for (int mt = 0; mt < 2; ++mt)
            #pragma unroll
            for (int nt = 0; nt < 4; ++nt)
                acc[mt][nt] = __builtin_amdgcn_wmma_f32_16x16x32_bf16(
                    false, af[mt].v, false, bfg[nt].v,
                    (short)0, acc[mt][nt], false, false);

#if HAS_ASYNC
        WAIT_ASYNC();          // my async writes into nxt are done
#else
        if (s + 1 < nsteps) {  // store prefetched regs into nxt
            uint4* qa = (uint4*)&sA[nxt][lofs]; qa[0] = ra0; qa[1] = ra1;
            uint4* qw = (uint4*)&sB[nxt][lofs]; qw[0] = rw0; qw[1] = rw1;
        }
#endif
        __syncthreads();       // everyone done writing nxt / reading cur
    }

    // ---- epilogue: C/D layout = lane(0..15)=N col, lane>>4 picks M rows v / v+8 ----
    const int colBase = nbase + wn * 64;
    const int rowBase = mbase + wm * 32;
    const int nloc = lane & 15;
    const int mofs = (lane >> 4) << 3;

    #pragma unroll
    for (int mt = 0; mt < 2; ++mt) {
        #pragma unroll
        for (int nt = 0; nt < 4; ++nt) {
            const int col = colBase + nt * 16 + nloc;
            if (col >= N) continue;
            const float bv = bias[col];
            #pragma unroll
            for (int v = 0; v < 8; ++v) {
                const int row = rowBase + mt * 16 + mofs + v;
                float val = acc[mt][nt][v] + bv;
                if (EPI == 0) {
                    outF[(size_t)row * ldOutF + col] = val;
                } else if (EPI == 1) {
                    outB[(size_t)row * ldOutB + col] = f32_to_bf16(tanhf(val));
                } else if (EPI == 2) {
                    float hd = aux0[(size_t)row * ldAux0 + col] + val * dt[row];
                    outF[(size_t)row * ldOutF + col] = hd;
                    outB[(size_t)row * ldOutB + col] = f32_to_bf16(hd);
                } else { // EPI == 3
                    float o = (ytype[col] == 0) ? val : aux0[(size_t)row * ldAux0 + col];
                    outF[(size_t)row * ldOutF + col] = o;
                }
            }
        }
    }
}

// ---------------- elementwise helpers ----------------
// fp32 weight (rows x cols) -> bf16, zero-padded to padRows
__global__ void __launch_bounds__(256)
pack_weight_kernel(const float* __restrict__ src, unsigned short* __restrict__ dst,
                   int rows, int cols, int padRows)
{
    int idx = blockIdx.x * 256 + threadIdx.x;
    if (idx >= padRows * cols) return;
    int r = idx / cols;
    float v = (r < rows) ? src[idx] : 0.0f;
    dst[idx] = f32_to_bf16(v);
}

// Pack [ht | ti] -> bf16 (B x 1088)
__global__ void __launch_bounds__(256)
pack_hti_kernel(const float* __restrict__ state, const float* __restrict__ ti,
                unsigned short* __restrict__ dst)
{
    int idx = blockIdx.x * 256 + threadIdx.x;
    if (idx >= BATCH * HTI) return;
    int m = idx / HTI;
    int c = idx - m * HTI;
    float v = (c < KH) ? state[(size_t)m * STLD + KOUT + c]
                       : ti[(size_t)m * KT + (c - KH)];
    dst[idx] = f32_to_bf16(v);
}

// xt fp32 -> bf16 (B x 512)
__global__ void __launch_bounds__(256)
pack_xt_kernel(const float* __restrict__ xt, unsigned short* __restrict__ dst)
{
    int idx = blockIdx.x * 256 + threadIdx.x;
    if (idx >= BATCH * KXT) return;
    dst[idx] = f32_to_bf16(xt[idx]);
}

// GRU gate combine: h_new = (1-z)*n + z*ht_dt -> d_out[:,64:] (f32) + bf16 copy
__global__ void __launch_bounds__(256)
gru_kernel(const float* __restrict__ gi, const float* __restrict__ gh,
           const float* __restrict__ htdt,
           float* __restrict__ out, unsigned short* __restrict__ hnewB)
{
    int idx = blockIdx.x * 256 + threadIdx.x;
    if (idx >= BATCH * KH) return;
    int m = idx >> 10;          // /1024
    int n = idx & (KH - 1);
    size_t base = (size_t)m * (3 * KH) + n;
    float ir = gi[base], iz = gi[base + KH], in_ = gi[base + 2 * KH];
    float hr = gh[base], hz = gh[base + KH], hn  = gh[base + 2 * KH];
    float r = sigmoidf(ir + hr);
    float z = sigmoidf(iz + hz);
    float nn = tanhf(in_ + r * hn);
    float hd = htdt[idx];
    float hnew = (1.0f - z) * nn + z * hd;
    out[(size_t)m * STLD + KOUT + n] = hnew;
    hnewB[idx] = f32_to_bf16(hnew);
}

// ---------------- workspace layout (bytes) ----------------
#define OFF_HTI    ((size_t)0)                                   // B*1088 bf16
#define OFF_ACT1   (OFF_HTI   + (size_t)BATCH * HTI   * 2)       // B*2176 bf16
#define OFF_XT     (OFF_ACT1  + (size_t)BATCH * FHID  * 2)       // B*512  bf16
#define OFF_HTDTF  (OFF_XT    + (size_t)BATCH * KXT   * 2)       // B*1024 f32
#define OFF_HTDTB  (OFF_HTDTF + (size_t)BATCH * KH    * 4)       // B*1024 bf16
#define OFF_GI     (OFF_HTDTB + (size_t)BATCH * KH    * 2)       // B*3072 f32
#define OFF_GH     (OFF_GI    + (size_t)BATCH * 3*KH  * 4)       // B*3072 f32
#define OFF_HNEWB  (OFF_GH    + (size_t)BATCH * 3*KH  * 4)       // B*1024 bf16
#define OFF_W1B    (OFF_HNEWB + (size_t)BATCH * KH    * 2)       // 2176x1088 bf16
#define OFF_W2B    (OFF_W1B   + (size_t)FHID * HTI    * 2)       // 1024x2176 bf16
#define OFF_WIHB   (OFF_W2B   + (size_t)KH * FHID     * 2)       // 3072x512  bf16
#define OFF_WHHB   (OFF_WIHB  + (size_t)(3*KH) * KXT  * 2)       // 3072x1024 bf16
#define OFF_LYB    (OFF_WHHB  + (size_t)(3*KH) * KH   * 2)       // 128x1024  bf16 (N padded 64->128)

extern "C" void kernel_launch(void* const* d_in, const int* in_sizes, int n_in,
                              void* d_out, int out_size, void* d_ws, size_t ws_size,
                              hipStream_t stream)
{
    const float* state  = (const float*)d_in[0];
    const float* xt     = (const float*)d_in[1];
    const float* dt     = (const float*)d_in[2];
    const float* ti     = (const float*)d_in[3];
    const float* f_W1   = (const float*)d_in[4];
    const float* f_b1   = (const float*)d_in[5];
    const float* f_W2   = (const float*)d_in[6];
    const float* f_b2   = (const float*)d_in[7];
    const float* W_ih   = (const float*)d_in[8];
    const float* W_hh   = (const float*)d_in[9];
    const float* b_ih   = (const float*)d_in[10];
    const float* b_hh   = (const float*)d_in[11];
    const float* Ly_W   = (const float*)d_in[12];
    const float* Ly_b   = (const float*)d_in[13];
    const int*   y_type = (const int*)d_in[14];
    float* out = (float*)d_out;

    char* ws = (char*)d_ws;
    unsigned short* hti    = (unsigned short*)(ws + OFF_HTI);
    unsigned short* act1   = (unsigned short*)(ws + OFF_ACT1);
    unsigned short* xtB    = (unsigned short*)(ws + OFF_XT);
    float*          htdtF  = (float*)(ws + OFF_HTDTF);
    unsigned short* htdtB  = (unsigned short*)(ws + OFF_HTDTB);
    float*          gi     = (float*)(ws + OFF_GI);
    float*          gh     = (float*)(ws + OFF_GH);
    unsigned short* hnewB  = (unsigned short*)(ws + OFF_HNEWB);
    unsigned short* w1B    = (unsigned short*)(ws + OFF_W1B);
    unsigned short* w2B    = (unsigned short*)(ws + OFF_W2B);
    unsigned short* wihB   = (unsigned short*)(ws + OFF_WIHB);
    unsigned short* whhB   = (unsigned short*)(ws + OFF_WHHB);
    unsigned short* lyB    = (unsigned short*)(ws + OFF_LYB);

    const dim3 blk(256);
    const int mblocks = BATCH / BM;   // 64
    auto eb = [](int n) { return dim3((n + 255) / 256); };

    // 0) pre-pack weights to bf16 (Ly_W padded in N to 128 with zeros)
    pack_weight_kernel<<<eb(FHID * HTI),     blk, 0, stream>>>(f_W1, w1B, FHID,   HTI,  FHID);
    pack_weight_kernel<<<eb(KH * FHID),      blk, 0, stream>>>(f_W2, w2B, KH,     FHID, KH);
    pack_weight_kernel<<<eb(3 * KH * KXT),   blk, 0, stream>>>(W_ih, wihB, 3*KH,  KXT,  3*KH);
    pack_weight_kernel<<<eb(3 * KH * KH),    blk, 0, stream>>>(W_hh, whhB, 3*KH,  KH,   3*KH);
    pack_weight_kernel<<<eb(BN * KH),        blk, 0, stream>>>(Ly_W, lyB,  KOUT,  KH,   BN);

    // 1) pack [ht | ti] and xt to bf16
    pack_hti_kernel<<<eb(BATCH * HTI), blk, 0, stream>>>(state, ti, hti);
    pack_xt_kernel <<<eb(BATCH * KXT), blk, 0, stream>>>(xt, xtB);

    // 2) GEMM1: tanh(h_ti @ f_W1^T + f_b1) -> act1 (bf16)
    gemm_bf16_wmma<1><<<dim3(FHID / BN, mblocks), blk, 0, stream>>>(
        hti, w1B, f_b1, FHID, HTI,
        nullptr, 0, act1, FHID, nullptr, 0, nullptr, nullptr);

    // 3) GEMM2: ht_dt = ht + (act1 @ f_W2^T + f_b2) * dt  -> f32 + bf16
    gemm_bf16_wmma<2><<<dim3(KH / BN, mblocks), blk, 0, stream>>>(
        act1, w2B, f_b2, KH, FHID,
        htdtF, KH, htdtB, KH, state + KOUT, STLD, dt, nullptr);

    // 4) GEMM3: gi = xt @ W_ih^T + b_ih
    gemm_bf16_wmma<0><<<dim3((3 * KH) / BN, mblocks), blk, 0, stream>>>(
        xtB, wihB, b_ih, 3 * KH, KXT,
        gi, 3 * KH, nullptr, 0, nullptr, 0, nullptr, nullptr);

    // 5) GEMM4: gh = ht_dt @ W_hh^T + b_hh
    gemm_bf16_wmma<0><<<dim3((3 * KH) / BN, mblocks), blk, 0, stream>>>(
        htdtB, whhB, b_hh, 3 * KH, KH,
        gh, 3 * KH, nullptr, 0, nullptr, 0, nullptr, nullptr);

    // 6) GRU combine -> h_new into d_out[:,64:] (f32) + bf16 copy
    gru_kernel<<<eb(BATCH * KH), blk, 0, stream>>>(gi, gh, htdtF, out, hnewB);

    // 7) GEMM5: Ly_out with y_type mask -> d_out[:,0:64] (weights N-padded, loads unguarded)
    gemm_bf16_wmma<3><<<dim3(1, mblocks), blk, 0, stream>>>(
        hnewB, lyB, Ly_b, KOUT, KH,
        out, STLD, nullptr, 0, state, STLD, nullptr, y_type);
}
